// BiLSTM_CRF_81612968559236
// MI455X (gfx1250) — matrix operand order repaired and checked by reading, not compiled, not run on hardware
//
#include <hip/hip_runtime.h>

typedef unsigned short u16;
typedef unsigned int   u32;
typedef __attribute__((ext_vector_type(16))) __bf16 bf16x16;
typedef __attribute__((ext_vector_type(8)))  float   floatx8;
typedef __attribute__((ext_vector_type(4)))  u32     u32x4;

#define Ee 512
#define H2 512
#define G4 2048
#define Bb 64
#define Ss 512
#define MM (Ss*Bb)   // 32768 rows
#define TGP 16       // padded tag dim

union FragBits { u32x4 q[2]; bf16x16 v; };

__device__ __forceinline__ u16 f2bf(float f){
  u32 u = __float_as_uint(f);
  u32 r = (u + 0x7FFFu + ((u >> 16) & 1u)) >> 16;   // RNE
  return (u16)r;
}
__device__ __forceinline__ float bf2f(u16 h){ return __uint_as_float(((u32)h) << 16); }

// A fragment (16x32 bf16, MxK): lane<16 -> row=lane, K 0..15;
// lane>=16 -> row=lane-16, K 16..31. Two contiguous 16B loads.
__device__ __forceinline__ bf16x16 load_a(const u16* p0, int ld, int lane){
  int half = lane >> 4, row = lane & 15;
  const u16* p = p0 + row * ld + half * 16;
  FragBits f; f.q[0] = *(const u32x4*)p; f.q[1] = *(const u32x4*)(p + 8);
  return f.v;
}
// B fragment (32x16 bf16, KxN) from row-major W[n][k] (B[k][n] = W[n][k]):
// lane col = lane&15, K range = (lane>>4)*16 .. +15 -> one contiguous 32B run.
__device__ __forceinline__ bf16x16 load_b(const u16* p0, int ld, int lane){
  int col = lane & 15, half = lane >> 4;
  const u16* p = p0 + col * ld + half * 16;
  FragBits f; f.q[0] = *(const u32x4*)p; f.q[1] = *(const u32x4*)(p + 8);
  return f.v;
}
__device__ __forceinline__ floatx8 wmma_bf16(bf16x16 a, bf16x16 b, floatx8 c){
  return __builtin_amdgcn_wmma_f32_16x16x32_bf16(false, a, false, b, (short)0, c, false, false);
}
__device__ __forceinline__ float sigm(float x){ return 1.f / (1.f + __expf(-x)); }

// ---------------- weight casts ----------------
__global__ void k_cast(u16* dst, const float* src, int n){
  int i = blockIdx.x * blockDim.x + threadIdx.x;
  if (i < n) dst[i] = f2bf(src[i]);
}
__global__ void k_cast_wout(u16* dst, const float* src){
  int i = blockIdx.x * blockDim.x + threadIdx.x;    // 16*1024
  if (i < TGP * 1024){
    int t = i >> 10, k = i & 1023;
    dst[i] = (t < 12) ? f2bf(src[t * 1024 + k]) : (u16)0;
  }
}
// ---------------- embedding gather ----------------
__global__ void k_embed(u16* xbf, const int* sent, const float* emb){
  int i = blockIdx.x * blockDim.x + threadIdx.x;    // < 32768*512
  if (i >= MM * Ee) return;
  int e = i & 511, row = i >> 9;
  int s = row >> 6, b = row & 63;
  int tok = sent[b * Ss + s];
  xbf[i] = f2bf(emb[tok * Ee + e]);
}
// ---------------- input-projection GEMM: Gin[dir][m][n] = X@W_ih^T + b ----------------
__global__ void k_gin(const u16* __restrict__ xbf, const u16* __restrict__ wih,
                      const float* __restrict__ bias_f, const float* __restrict__ bias_b,
                      u16* __restrict__ gin){
  int lane = threadIdx.x & 31, wave = threadIdx.x >> 5;
  int m0 = blockIdx.x * 16;
  int n0 = blockIdx.y * 64 + wave * 16;
  int dir = blockIdx.z;
  const u16* A = xbf + (size_t)m0 * Ee;
  const u16* W = wih + (size_t)dir * G4 * Ee + (size_t)n0 * Ee;
  floatx8 acc = {};
  #pragma unroll 4
  for (int k0 = 0; k0 < Ee; k0 += 32){
    bf16x16 a = load_a(A + k0, Ee, lane);
    bf16x16 b = load_b(W + k0, Ee, lane);
    acc = wmma_bf16(a, b, acc);
  }
  int col = lane & 15, half = lane >> 4;
  const float* bias = dir ? bias_b : bias_f;
  float bv = bias[n0 + col];
  u16* out = gin + (size_t)dir * MM * G4 + (size_t)m0 * G4 + n0 + col;
  #pragma unroll
  for (int r = 0; r < 8; ++r)
    out[(size_t)(r + 8 * half) * G4] = f2bf(acc[r] + bv);
}
// ---------------- sequential LSTM recurrence (both dirs, batch-tiled) ----------------
// Double-buffers the next timestep's 64KB Gin tile into LDS with
// GLOBAL_LOAD_ASYNC_TO_LDS_B128 while the current step's WMMAs execute.
__global__ void __launch_bounds__(512) k_lstm(const u16* __restrict__ gin,
                                              const u16* __restrict__ whh,
                                              u16* __restrict__ hs){
  __shared__ u16 h_sh[16 * H2];              // h_prev, bf16, [m][k]       (16 KB)
  __shared__ u16 gin_sh[2 * 16 * G4];        // async staging, 2 buffers  (128 KB)
  int b0 = blockIdx.x * 16;
  int dir = blockIdx.y;
  int tid = threadIdx.x;
  int lane = tid & 31, wave = tid >> 5;
  int col = lane & 15, half = lane >> 4;
  const u16* W    = whh + (size_t)dir * G4 * H2;
  const u16* GinD = gin + (size_t)dir * MM * G4;
  // LDS aperture maps flat addr[31:0] -> LDS byte offset
  unsigned ginLds = (unsigned)(size_t)(void*)gin_sh;

  for (int i = tid; i < 16 * H2; i += 512) h_sh[i] = 0;
  float cst[2][8];
  #pragma unroll
  for (int jt = 0; jt < 2; ++jt)
    #pragma unroll
    for (int r = 0; r < 8; ++r) cst[jt][r] = 0.f;

  { // prologue: stage step 0 into buffer 0 (64KB contiguous, 8 x b128 per thread)
    int s0 = dir ? (Ss - 1) : 0;
    const char* g = (const char*)(GinD + (size_t)(s0 * Bb + b0) * G4);
    #pragma unroll
    for (int r = 0; r < 8; ++r){
      unsigned off = (unsigned)((tid + r * 512) * 16);
      asm volatile("global_load_async_to_lds_b128 %0, %1, off"
                   :: "v"(ginLds + off), "v"(g + off) : "memory");
    }
  }
  asm volatile("s_wait_asynccnt 0x0" ::: "memory");
  __syncthreads();

  for (int t = 0; t < Ss; ++t){
    int s = dir ? (Ss - 1 - t) : t;
    int buf = t & 1;
    if (t + 1 < Ss){                          // kick off next step's staging now
      int sn = dir ? (Ss - 2 - t) : (t + 1);
      const char* g = (const char*)(GinD + (size_t)(sn * Bb + b0) * G4);
      unsigned dst = ginLds + (unsigned)((buf ^ 1) * 16 * G4 * 2);
      #pragma unroll
      for (int r = 0; r < 8; ++r){
        unsigned off = (unsigned)((tid + r * 512) * 16);
        asm volatile("global_load_async_to_lds_b128 %0, %1, off"
                     :: "v"(dst + off), "v"(g + off) : "memory");
      }
    }
    const u16* ginRow = gin_sh + buf * 16 * G4;
    u16 hreg[2][8];
    #pragma unroll
    for (int jt = 0; jt < 2; ++jt){
      int ncol = wave * 32 + jt * 16;         // hidden-column base this wave owns
      floatx8 ai, af, ag, ao;
      #pragma unroll
      for (int r = 0; r < 8; ++r){            // C init = staged x-projection (LDS)
        const u16* gp = ginRow + (r + 8 * half) * G4 + ncol + col;
        ai[r] = bf2f(gp[0]);    af[r] = bf2f(gp[512]);
        ag[r] = bf2f(gp[1024]); ao[r] = bf2f(gp[1536]);
      }
      const u16* Wi = W + (size_t)(ncol)        * H2;
      const u16* Wf = W + (size_t)(512  + ncol) * H2;
      const u16* Wg = W + (size_t)(1024 + ncol) * H2;
      const u16* Wo = W + (size_t)(1536 + ncol) * H2;
      #pragma unroll 2
      for (int k0 = 0; k0 < H2; k0 += 32){
        bf16x16 a = load_a(&h_sh[k0], H2, lane); // shared A across 4 gate GEMMs
        ai = wmma_bf16(a, load_b(Wi + k0, H2, lane), ai);
        af = wmma_bf16(a, load_b(Wf + k0, H2, lane), af);
        ag = wmma_bf16(a, load_b(Wg + k0, H2, lane), ag);
        ao = wmma_bf16(a, load_b(Wo + k0, H2, lane), ao);
      }
      // i/f/g/o co-resident per lane/slot: cell update fully in registers
      #pragma unroll
      for (int r = 0; r < 8; ++r){
        float cn = sigm(af[r]) * cst[jt][r] + sigm(ai[r]) * tanhf(ag[r]);
        cst[jt][r] = cn;
        hreg[jt][r] = f2bf(sigm(ao[r]) * tanhf(cn));
      }
    }
    __syncthreads();                          // all WMMA reads of h_prev done
    u16* hOut = hs + ((size_t)(dir * Ss + s) * Bb + b0) * H2;
    #pragma unroll
    for (int jt = 0; jt < 2; ++jt){
      int jcol = wave * 32 + jt * 16 + col;
      #pragma unroll
      for (int r = 0; r < 8; ++r){
        int mrow = r + 8 * half;
        h_sh[mrow * H2 + jcol] = hreg[jt][r];
        hOut[(size_t)mrow * H2 + jcol] = hreg[jt][r];
      }
    }
    asm volatile("s_wait_asynccnt 0x0" ::: "memory"); // own staging chunks landed
    __syncthreads();                          // all waves' chunks + new h visible
  }
}
// ---------------- feats GEMM: [32768 x 1024] @ WoutT[1024 x 16] ----------------
__global__ void k_feats(const u16* __restrict__ hs, const u16* __restrict__ wout,
                        const float* __restrict__ bout, float* __restrict__ feats){
  int lane = threadIdx.x & 31;
  int m0 = blockIdx.x * 16;
  floatx8 acc = {};
  #pragma unroll 4
  for (int kt = 0; kt < 32; ++kt){
    int dir = kt >> 4;
    int k0 = (kt & 15) * 32;
    const u16* A = hs + (size_t)dir * Ss * Bb * H2 + (size_t)m0 * H2 + k0;
    bf16x16 a = load_a(A, H2, lane);
    bf16x16 b = load_b(wout + kt * 32, 1024, lane);
    acc = wmma_bf16(a, b, acc);
  }
  int col = lane & 15, half = lane >> 4;
  float bv = (col < 12) ? bout[col] : 0.f;
  #pragma unroll
  for (int r = 0; r < 8; ++r)
    feats[(size_t)(m0 + r + 8 * half) * TGP + col] = acc[r] + bv;
}
// ---------------- Viterbi (Tg=12, per batch) ----------------
__global__ void k_viterbi(const float* __restrict__ feats, const float* __restrict__ trans,
                          unsigned char* __restrict__ bps,
                          float* __restrict__ out_scores, float* __restrict__ out_paths){
  int b = blockIdx.x, t = threadIdx.x;
  __shared__ float fv[16];
  __shared__ float tr[144];
  __shared__ float term[16];
  for (int i = t; i < 144; i += 32) tr[i] = trans[i];
  if (t < 16) fv[t] = (t == 10) ? 0.f : -10000.f;   // START=10
  __syncthreads();
  for (int s = 0; s < Ss; ++s){
    float best = -3.4e38f; int bp = 0; float ft = 0.f;
    if (t < 12){
      #pragma unroll
      for (int p = 0; p < 12; ++p){
        float v = fv[p] + tr[t * 12 + p];
        if (v > best){ best = v; bp = p; }
      }
      ft = feats[(size_t)(s * Bb + b) * TGP + t];
    }
    __syncthreads();
    if (t < 12){
      fv[t] = best + ft;
      bps[((size_t)b * Ss + s) * 12 + t] = (unsigned char)bp;
    }
    __syncthreads();
  }
  if (t < 12) term[t] = fv[t] + tr[11 * 12 + t];    // STOP=11
  __syncthreads();
  if (t == 0){
    int bt = 0; float bs = term[0];
    for (int p = 1; p < 12; ++p) if (term[p] > bs){ bs = term[p]; bt = p; }
    out_scores[b] = bs;
    int tag = bt;
    for (int s = Ss - 1; s >= 0; --s){
      out_paths[(size_t)b * Ss + s] = (float)tag;
      tag = bps[((size_t)b * Ss + s) * 12 + tag];
    }
  }
}

extern "C" void kernel_launch(void* const* d_in, const int* in_sizes, int n_in,
                              void* d_out, int out_size, void* d_ws, size_t ws_size,
                              hipStream_t stream){
  const int*   sent  = (const int*)  d_in[0];
  const float* emb   = (const float*)d_in[1];
  const float* Wih_f = (const float*)d_in[2];
  const float* Whh_f = (const float*)d_in[3];
  const float* b_f   = (const float*)d_in[4];
  const float* Wih_b = (const float*)d_in[5];
  const float* Whh_b = (const float*)d_in[6];
  const float* b_b   = (const float*)d_in[7];
  const float* Wout  = (const float*)d_in[8];
  const float* bout  = (const float*)d_in[9];
  const float* trans = (const float*)d_in[10];

  char* ws = (char*)d_ws;
  u16* xbf    = (u16*)(ws + 0);              //  32 MB  bf16 embedded inputs [s][b][e]
  u16* wih    = (u16*)(ws + 33554432);       //   4 MB  bf16 W_ih (f,b)
  u16* whh    = (u16*)(ws + 37748736);       //   4 MB  bf16 W_hh (f,b)
  u16* woutb  = (u16*)(ws + 41943040);       //  32 KB  bf16 W_out padded to 16 rows
  u16* gin    = (u16*)(ws + 41975808);       // 256 MB  bf16 gate pre-activations
  u16* hsbuf  = (u16*)(ws + 310411264);      //  64 MB  bf16 hidden states [dir][s][b][h]
  float* feats = (float*)(ws + 377520128);   //   2 MB  fp32 feats [m][16]
  unsigned char* bps = (unsigned char*)(ws + 379617280); // 384 KB backpointers

  const int nW = G4 * Ee;                    // 2048*512
  k_cast<<<(nW + 255) / 256, 256, 0, stream>>>(wih,      Wih_f, nW);
  k_cast<<<(nW + 255) / 256, 256, 0, stream>>>(wih + nW, Wih_b, nW);
  k_cast<<<(nW + 255) / 256, 256, 0, stream>>>(whh,      Whh_f, nW);
  k_cast<<<(nW + 255) / 256, 256, 0, stream>>>(whh + nW, Whh_b, nW);
  k_cast_wout<<<64, 256, 0, stream>>>(woutb, Wout);
  k_embed<<<(MM * Ee) / 256, 256, 0, stream>>>(xbf, sent, emb);
  k_gin<<<dim3(MM / 16, G4 / 64, 2), 128, 0, stream>>>(xbf, wih, b_f, b_b, gin);
  k_lstm<<<dim3(Bb / 16, 2), 512, 0, stream>>>(gin, whh, hsbuf);
  k_feats<<<MM / 16, 32, 0, stream>>>(hsbuf, woutb, bout, feats);
  k_viterbi<<<Bb, 32, 0, stream>>>(feats, trans, bps, (float*)d_out, (float*)d_out + Bb);
}
